// AttentionCell_8280696946870
// MI455X (gfx1250) — matrix operand order repaired and checked by reading, not compiled, not run on hardware
//
#include <hip/hip_runtime.h>

// ---------------------------------------------------------------------------
// AttentionCell for MI455X (gfx1250, wave32, WMMA).
// - bf16 V_WMMA_F32_16X16X32_BF16 everywhere (problem is HBM-bound: 2x 268MB
//   reads of feats; bf16 matrix compute sits far below the 23.3 TB/s floor).
// - Weights pre-converted to bf16 once -> hot loops do raw 16B copies.
// - Score GEMM fused with +hid_proj, tanh, *w_score, N-reduction (never
//   materializes the 134MB feats_proj tensor), double-buffered LDS staging,
//   async global->LDS copies for the weight tile when the builtin exists.
// ---------------------------------------------------------------------------

typedef __attribute__((ext_vector_type(16))) __bf16 v16bf;
typedef __attribute__((ext_vector_type(8)))  float  v8f;
typedef int v4i __attribute__((vector_size(16)));

union FragU { uint4 q[2]; v16bf v; };

__device__ __forceinline__ unsigned short f2bf(float f) {
    unsigned int u = __float_as_uint(f);
    u += 0x7FFFu + ((u >> 16) & 1u);          // RNE
    return (unsigned short)(u >> 16);
}

__device__ __forceinline__ unsigned int pack2_bf16(float x, float y) {
#if __has_builtin(__builtin_amdgcn_cvt_pk_bf16_f32)
    auto v = __builtin_amdgcn_cvt_pk_bf16_f32(x, y);   // v_cvt_pk_bf16_f32
    return __builtin_bit_cast(unsigned int, v);
#else
    return (unsigned int)f2bf(x) | ((unsigned int)f2bf(y) << 16);
#endif
}

// ---- async global->LDS 16B copy (CDNA5 GLOBAL_LOAD_ASYNC_TO_LDS_B128) -----
#if __has_builtin(__builtin_amdgcn_global_load_async_to_lds_b128) && \
    __has_builtin(__builtin_amdgcn_s_wait_asynccnt)
#define HAVE_ASYNC_COPY 1
#else
#define HAVE_ASYNC_COPY 0
#endif

__device__ __forceinline__ void copy16_g2l(const unsigned short* gsrc, unsigned short* ldst) {
#if HAVE_ASYNC_COPY
    typedef __attribute__((address_space(1))) v4i* GP;   // global v4i*
    typedef __attribute__((address_space(3))) v4i* LP;   // LDS v4i*
    __builtin_amdgcn_global_load_async_to_lds_b128((GP)gsrc, (LP)ldst, 0, 0);
#else
    *(uint4*)ldst = *(const uint4*)gsrc;
#endif
}
__device__ __forceinline__ void wait_async_copies() {
#if HAVE_ASYNC_COPY
    __builtin_amdgcn_s_wait_asynccnt(0);
#endif
}

// A-matrix 16x32 bf16 fragment (CDNA5 wave32 layout):
// lanes 0-15 : M=lane,    K = {0..7, 16..23}
// lanes 16-31: M=lane-16, K = {8..15, 24..31}
// LDS tile [rows][40 halfs]: 80B stride -> conflict-free, 16B-aligned chunks.
__device__ __forceinline__ v16bf load_a_frag(const unsigned short* At, int m_base, int lane) {
    int half = lane >> 4, mr = lane & 15;
    const char* base = (const char*)(At + (m_base + mr) * 40);
    FragU f;
    f.q[0] = *(const uint4*)(base + 16 * half);
    f.q[1] = *(const uint4*)(base + 16 * half + 32);
    return f.v;
}
// B-matrix 32x16: lane n<16 holds column n K={0..7,16..23}; lane n+16 K={8..15,24..31}.
__device__ __forceinline__ v16bf load_b_frag(const unsigned short* Bt, int n_base, int lane) {
    int half = lane >> 4, nr = lane & 15;
    const char* base = (const char*)(Bt + (n_base + nr) * 40);
    FragU f;
    f.q[0] = *(const uint4*)(base + 16 * half);
    f.q[1] = *(const uint4*)(base + 16 * half + 32);
    return f.v;
}

// ---------------------------------------------------------------------------
// fp32 -> packed bf16 weight conversion (runs once per weight matrix)
// ---------------------------------------------------------------------------
__global__ __launch_bounds__(256) void cvt_bf16_kernel(
    const float* __restrict__ src, unsigned int* __restrict__ dst, int npairs)
{
    int i = blockIdx.x * 256 + threadIdx.x;
    if (i < npairs) {
        float2 f = *(const float2*)(src + 2 * (long)i);
        dst[i] = pack2_bf16(f.x, f.y);
    }
}

// ---------------------------------------------------------------------------
// C = A @ Wbf^T (+bias)(+=) ; A fp32 [M,lda], Wbf bf16 [N,ldw].
// 64x64 tile, 4 waves. M,N % 64 == 0, K % 32 == 0.
// ---------------------------------------------------------------------------
template <bool HAS_BIAS, bool ACCUM>
__global__ __launch_bounds__(128) void gemm_bf16_kernel(
    const float* __restrict__ A, int lda,
    const unsigned short* __restrict__ Wbf, int ldw,
    const float* __restrict__ bias,
    float* __restrict__ Cmat, int ldc, int K)
{
    __shared__ __align__(16) unsigned short At[64 * 40];
    __shared__ __align__(16) unsigned short Bt[64 * 40];
    const int tid = threadIdx.x;
    const int wave = tid >> 5, lane = tid & 31;
    const int m0 = blockIdx.y * 64, n0 = blockIdx.x * 64;

    v8f acc[4] = {};
    for (int kc = 0; kc < K; kc += 32) {
        __syncthreads();
        #pragma unroll
        for (int i = 0; i < 8; ++i) {                  // A: 64x32 fp32->bf16 pairs
            int idx = i * 128 + tid;
            int m = idx >> 4, kp = idx & 15;
            float2 f = *(const float2*)(A + (long)(m0 + m) * lda + kc + kp * 2);
            *(unsigned int*)&At[m * 40 + kp * 2] = pack2_bf16(f.x, f.y);
        }
        #pragma unroll
        for (int i = 0; i < 2; ++i) {                  // W: 64x32 bf16 raw 16B copies
            int idx = i * 128 + tid;
            int n = idx >> 2, g = idx & 3;
            copy16_g2l(Wbf + (long)(n0 + n) * ldw + kc + g * 8, &Bt[n * 40 + g * 8]);
        }
        wait_async_copies();
        __syncthreads();
        v16bf af = load_a_frag(At, wave * 16, lane);
        v16bf bfr[4];
        #pragma unroll
        for (int j = 0; j < 4; ++j) bfr[j] = load_b_frag(Bt, j * 16, lane);
        #pragma unroll
        for (int j = 0; j < 4; ++j)
            acc[j] = __builtin_amdgcn_wmma_f32_16x16x32_bf16(
                false, af, false, bfr[j], (short)0, acc[j], false, false);
    }
    const int hl = lane >> 4, nl = lane & 15;
    #pragma unroll
    for (int j = 0; j < 4; ++j) {
        #pragma unroll
        for (int i = 0; i < 8; ++i) {
            int row = m0 + wave * 16 + i + (hl ? 8 : 0);
            int col = n0 + j * 16 + nl;
            float v = acc[j][i];
            if constexpr (HAS_BIAS) v += bias[col];
            long off = (long)row * ldc + col;
            if constexpr (ACCUM) v += Cmat[off];
            Cmat[off] = v;
        }
    }
}

// ---------------------------------------------------------------------------
// Fused score: e[b,t] = sum_h tanh((feats@W_i2h^T)[t,b,h] + hid_proj[b,h]) * w_score[h]
// M = T*B = 131072 (row r = t*512+b), N = 256, K = 512.
// 8 waves: mw -> 16-row subtile, nw -> 128-col half. Double-buffered LDS.
// ---------------------------------------------------------------------------
__global__ __launch_bounds__(256) void attn_score_kernel(
    const float* __restrict__ feats,          // [T*B, 512] fp32
    const unsigned short* __restrict__ Wbf,   // [256, 512] bf16
    const float* __restrict__ hid_proj,       // [512, 256]
    const float* __restrict__ w_score,        // [256]
    float* __restrict__ e_bt)                 // [B, T]
{
    const int C = 512, H = 256, T = 256;
    __shared__ __align__(16) unsigned short At[2][64 * 40];    // 10 KB
    __shared__ __align__(16) unsigned short Bt[2][256 * 40];   // 40 KB
    __shared__ float ws_lds[256];
    __shared__ float e_red[64];

    const int tid = threadIdx.x;
    const int wave = tid >> 5, lane = tid & 31;
    const int mw = wave & 3, nw = wave >> 2;
    const long r0 = (long)blockIdx.x * 64;

    ws_lds[tid] = w_score[tid];
    if (tid < 64) e_red[tid] = 0.0f;

    auto stage = [&](int buf, int kc) {
        #pragma unroll
        for (int i = 0; i < 4; ++i) {                  // A: 64x32 fp32->bf16 pairs
            int idx = i * 256 + tid;
            int m = idx >> 4, kp = idx & 15;
            float2 f = *(const float2*)(feats + (r0 + m) * C + kc + kp * 2);
            *(unsigned int*)&At[buf][m * 40 + kp * 2] = pack2_bf16(f.x, f.y);
        }
        #pragma unroll
        for (int i = 0; i < 4; ++i) {                  // W: 256x32 bf16, async 16B copies
            int idx = i * 256 + tid;
            int n = idx >> 2, g = idx & 3;
            copy16_g2l(Wbf + (long)n * C + kc + g * 8, &Bt[buf][n * 40 + g * 8]);
        }
    };

    v8f acc[8] = {};
    stage(0, 0);
    for (int c = 0; c < 16; ++c) {
        wait_async_copies();          // my async writes to LDS[buf] done
        __syncthreads();              // everyone's staging of buf visible
        if (c + 1 < 16) stage((c + 1) & 1, (c + 1) * 32);  // overlap with WMMA
        const int buf = c & 1;
        v16bf af = load_a_frag(At[buf], mw * 16, lane);
        v16bf bfr[8];
        #pragma unroll
        for (int j = 0; j < 8; ++j)
            bfr[j] = load_b_frag(Bt[buf], nw * 128 + j * 16, lane);
        #pragma unroll
        for (int j = 0; j < 8; ++j)
            acc[j] = __builtin_amdgcn_wmma_f32_16x16x32_bf16(
                false, af, false, bfr[j], (short)0, acc[j], false, false);
    }
    __syncthreads();

    // Epilogue: tanh(acc + hid_proj[b,n]) * w_score[n], reduce over N=256.
    const int hl = lane >> 4, nl = lane & 15;
    float partial[8];
    #pragma unroll
    for (int i = 0; i < 8; ++i) partial[i] = 0.0f;
    #pragma unroll
    for (int j = 0; j < 8; ++j) {
        int n = nw * 128 + j * 16 + nl;
        float wsn = ws_lds[n];
        #pragma unroll
        for (int i = 0; i < 8; ++i) {
            long r = r0 + mw * 16 + i + (hl ? 8 : 0);
            int b = (int)(r & 511);
            partial[i] += tanhf(acc[j][i] + hid_proj[b * H + n]) * wsn;
        }
    }
    #pragma unroll
    for (int i = 0; i < 8; ++i) {      // butterfly over 16-lane half (same row)
        float p = partial[i];
        p += __shfl_xor(p, 1, 32);
        p += __shfl_xor(p, 2, 32);
        p += __shfl_xor(p, 4, 32);
        p += __shfl_xor(p, 8, 32);
        partial[i] = p;
    }
    if (nl == 0) {                     // 2 waves per row -> ds_add_f32 (commutative)
        #pragma unroll
        for (int i = 0; i < 8; ++i)
            atomicAdd(&e_red[mw * 16 + i + (hl ? 8 : 0)], partial[i]);
    }
    __syncthreads();
    if (tid < 64) {
        long r = r0 + tid;
        int b = (int)(r & 511);
        int t = (int)(r >> 9);
        e_bt[(long)b * T + t] = e_red[tid];
    }
}

// ---------------------------------------------------------------------------
// Per-batch softmax over T + context = sum_t alpha[t] * feats[t,b,:]
// ---------------------------------------------------------------------------
__global__ __launch_bounds__(256) void softmax_context_kernel(
    const float* __restrict__ e_bt,    // [B,T]
    const float* __restrict__ feats,   // [T,B,C]
    float* __restrict__ alpha_out,     // [B,T]
    float* __restrict__ context)       // [B,C]
{
    const int T = 256, Bdim = 512, C = 512;
    __shared__ float sm[256];
    __shared__ float alpha_lds[256];
    const int b = blockIdx.x, tid = threadIdx.x;

    float e = e_bt[(long)b * T + tid];
    sm[tid] = e; __syncthreads();
    for (int s = 128; s > 0; s >>= 1) {
        if (tid < s) sm[tid] = fmaxf(sm[tid], sm[tid + s]);
        __syncthreads();
    }
    float mx = sm[0]; __syncthreads();
    float p = __expf(e - mx);
    sm[tid] = p; __syncthreads();
    for (int s = 128; s > 0; s >>= 1) {
        if (tid < s) sm[tid] += sm[tid + s];
        __syncthreads();
    }
    float a = p / sm[0];
    alpha_lds[tid] = a;
    alpha_out[(long)b * T + tid] = a;
    __syncthreads();

    float acc0 = 0.f, acc1 = 0.f;
    #pragma unroll 4
    for (int t = 0; t < T; ++t) {
        const float* row = feats + ((long)t * Bdim + b) * C;
        int tp = t + 8 < T ? t + 8 : T - 1;
        __builtin_prefetch(feats + ((long)tp * Bdim + b) * C + tid, 0, 0);
        float at = alpha_lds[t];
        acc0 += at * row[tid];
        acc1 += at * row[tid + 256];
    }
    context[(long)b * C + tid]       = acc0;
    context[(long)b * C + tid + 256] = acc1;
}

// ---------------------------------------------------------------------------
// GRU gates elementwise.
// ---------------------------------------------------------------------------
__global__ __launch_bounds__(256) void gru_kernel(
    const float* __restrict__ gi, const float* __restrict__ gh,
    const float* __restrict__ prev_hidden, float* __restrict__ cur_hidden)
{
    int idx = blockIdx.x * 256 + threadIdx.x;
    int b = idx >> 8, h = idx & 255;
    const float* gib = gi + (long)b * 768;
    const float* ghb = gh + (long)b * 768;
    float r = 1.f / (1.f + __expf(-(gib[h]       + ghb[h])));
    float z = 1.f / (1.f + __expf(-(gib[h + 256] + ghb[h + 256])));
    float n = tanhf(gib[h + 512] + r * ghb[h + 512]);
    cur_hidden[idx] = (1.f - z) * n + z * prev_hidden[idx];
}

// ---------------------------------------------------------------------------
extern "C" void kernel_launch(void* const* d_in, const int* in_sizes, int n_in,
                              void* d_out, int out_size, void* d_ws, size_t ws_size,
                              hipStream_t stream) {
    (void)in_sizes; (void)n_in; (void)out_size; (void)ws_size;
    const float* prev_hidden = (const float*)d_in[0];
    const float* feats       = (const float*)d_in[1];
    const float* cur_emb     = (const float*)d_in[2];
    const float* W_i2h       = (const float*)d_in[3];
    const float* W_h2h       = (const float*)d_in[4];
    const float* b_h2h       = (const float*)d_in[5];
    const float* w_score     = (const float*)d_in[6];
    const float* W_ih        = (const float*)d_in[7];
    const float* W_hh        = (const float*)d_in[8];
    const float* b_ih        = (const float*)d_in[9];
    const float* b_hh        = (const float*)d_in[10];

    const int T = 256, B = 512, C = 512, H = 256, E = 128;

    // workspace: fp32 scratch (~5.25MB) then bf16 weights (~1.7MB)
    float* ws       = (float*)d_ws;
    float* hid_proj = ws;                    // B*H
    float* e_bt     = hid_proj + B * H;      // B*T
    float* context  = e_bt + B * T;          // B*C
    float* gi       = context + B * C;       // B*3H
    float* gh       = gi + B * 3 * H;        // B*3H
    unsigned short* bfb     = (unsigned short*)(gh + B * 3 * H);
    unsigned short* Wi2h_bf = bfb;                       // H*C      = 131072
    unsigned short* Wh2h_bf = Wi2h_bf + H * C;           // H*H      = 65536
    unsigned short* Wih_bf  = Wh2h_bf + H * H;           // 3H*(C+E) = 491520
    unsigned short* Whh_bf  = Wih_bf + 3 * H * (C + E);  // 3H*H     = 196608

    float* cur_hidden_out = (float*)d_out;          // [B,H]
    float* alpha_out      = cur_hidden_out + B * H; // [B,T]

    // 0) one-time weight conversion fp32 -> bf16 (packed pairs)
    cvt_bf16_kernel<<<dim3(H * C / 2 / 256), 256, 0, stream>>>(W_i2h, (unsigned int*)Wi2h_bf, H * C / 2);
    cvt_bf16_kernel<<<dim3(H * H / 2 / 256), 256, 0, stream>>>(W_h2h, (unsigned int*)Wh2h_bf, H * H / 2);
    cvt_bf16_kernel<<<dim3(3 * H * (C + E) / 2 / 256), 256, 0, stream>>>(W_ih, (unsigned int*)Wih_bf, 3 * H * (C + E) / 2);
    cvt_bf16_kernel<<<dim3(3 * H * H / 2 / 256), 256, 0, stream>>>(W_hh, (unsigned int*)Whh_bf, 3 * H * H / 2);

    // 1) hid_proj = prev_hidden @ W_h2h^T + b_h2h
    gemm_bf16_kernel<true, false><<<dim3(H / 64, B / 64), 128, 0, stream>>>(
        prev_hidden, H, Wh2h_bf, H, b_h2h, hid_proj, H, H);

    // 2) fused score GEMM -> e[B,T]  (34 GFLOP, streams 268MB of feats)
    attn_score_kernel<<<dim3(T * B / 64), 256, 0, stream>>>(
        feats, Wi2h_bf, hid_proj, w_score, e_bt);

    // 3) softmax over T + context; alpha straight to d_out
    softmax_context_kernel<<<dim3(B), 256, 0, stream>>>(
        e_bt, feats, alpha_out, context);

    // 4) gi = [context|emb] @ W_ih^T + b_ih  (split-K over the concat)
    gemm_bf16_kernel<true, false><<<dim3(3 * H / 64, B / 64), 128, 0, stream>>>(
        context, C, Wih_bf, C + E, b_ih, gi, 3 * H, C);
    gemm_bf16_kernel<false, true><<<dim3(3 * H / 64, B / 64), 128, 0, stream>>>(
        cur_emb, E, Wih_bf + C, C + E, nullptr, gi, 3 * H, E);

    // 5) gh = prev_hidden @ W_hh^T + b_hh
    gemm_bf16_kernel<true, false><<<dim3(3 * H / 64, B / 64), 128, 0, stream>>>(
        prev_hidden, H, Whh_bf, H, b_hh, gh, 3 * H, H);

    // 6) GRU elementwise -> cur_hidden
    gru_kernel<<<dim3(B * H / 256), 256, 0, stream>>>(
        gi, gh, prev_hidden, cur_hidden_out);
}
